// MyIN_89601607729703
// MI455X (gfx1250) — matrix operand order repaired and checked by reading, not compiled, not run on hardware
//
#include <hip/hip_runtime.h>
#include <hip/hip_bf16.h>
#include <math.h>

// ---------------------------------------------------------------------------
// CDNA5 (gfx1250) interaction-network GNN, bf16 WMMA path.
// Every MLP layer (K<=64, N<=64) runs on v_wmma_f32_16x16x32_bf16 with f32
// accumulate. One wave = one 16-row tile; B fragments live in VGPRs; A
// fragments are staged per-wave in LDS and loaded in contiguous k order (k = 16*(lane/16) + e), same as B.
// ---------------------------------------------------------------------------

typedef __attribute__((ext_vector_type(16))) __bf16 v16bf;
typedef __attribute__((ext_vector_type(8)))  float  v8f;

#define WAVES 8   // 256 threads / wave32

// DS ops within a wave are processed in order (ISA ch.5 counters); the wait
// guarantees loaded data is in VGPRs and the barriers pin compiler ordering.
__device__ __forceinline__ void wave_sync_lds() {
  __builtin_amdgcn_wave_barrier();
  asm volatile("s_wait_dscnt 0" ::: "memory");
  __builtin_amdgcn_wave_barrier();
}

// B fragment (16-bit, 32x16, 8 VGPRs): lanes 0-15 hold K=kBase..kBase+15,
// lanes 16-31 hold K=kBase+16..kBase+31; N = nBase + lane%16. W is row-major
// [K][ld] bf16 in LDS.
__device__ __forceinline__ v16bf load_b_frag(const __bf16* W, int ld,
                                             int kBase, int nBase, int lane) {
  v16bf f;
  const int col = nBase + (lane & 15);
  const int k0  = kBase + ((lane >> 4) << 4);
#pragma unroll
  for (int e = 0; e < 16; ++e) f[e] = W[(k0 + e) * ld + col];
  return f;
}

// A fragment (16-bit, 16x32, 8 VGPRs): lane<16: K=0..15,
// lane>=16: K=16..31; M = lane%16. S row-major [16][ld] bf16 in LDS.
__device__ __forceinline__ v16bf load_a_frag(const __bf16* S, int ld,
                                             int kBase, int lane) {
  v16bf a;
  const int row = lane & 15;
  const int off = kBase + ((lane >> 4) << 4);
#pragma unroll
  for (int e = 0; e < 8; ++e) a[e] = S[row * ld + off + e];
#pragma unroll
  for (int e = 0; e < 8; ++e) a[8 + e] = S[row * ld + off + 8 + e];
  return a;
}

__global__ void zero_f32(float* __restrict__ p, int n) {
  int i = blockIdx.x * blockDim.x + threadIdx.x;
  if (i < n) p[i] = 0.0f;
}

// One fused 3-layer MLP over `rows` rows. Input row = concat of up to three
// gathered sources (widths WA/WB/WC, padded with zeros to K=32). Hidden 64,64
// with ReLU; output width OUT_W (<=16). Optional sigmoid, dense store, and
// atomic scatter-add (segment_sum) keyed by idxS.
template <int WA, int WB, int WC, int OUT_W, bool SIGMOID, bool STORE, bool SCATTER>
__global__ __launch_bounds__(256) void mlp_tile_kernel(
    const float* __restrict__ srcA, const long long* __restrict__ idxA,
    const float* __restrict__ srcB, const long long* __restrict__ idxB,
    const float* __restrict__ srcC, const long long* __restrict__ idxC,
    const float* __restrict__ w0, const float* __restrict__ b0,
    const float* __restrict__ w1, const float* __restrict__ b1,
    const float* __restrict__ w2, const float* __restrict__ b2,
    float* __restrict__ out, float* __restrict__ agg,
    const long long* __restrict__ idxS, int rows) {
  constexpr int KIN = WA + WB + WC;

  __shared__ __align__(16) __bf16 sW0[32 * 64];
  __shared__ __align__(16) __bf16 sW1[64 * 64];
  __shared__ __align__(16) __bf16 sW2[64 * 16];
  __shared__ float sB0[64], sB1[64], sB2[16];
  __shared__ __align__(16) __bf16 sStage[WAVES][16 * 64];

  const int tid = threadIdx.x;
  // Stage weights f32 -> bf16 into LDS (zero-pad K to 32 and N to 16).
  for (int i = tid; i < 32 * 64; i += 256) {
    const int k = i >> 6, n = i & 63;
    sW0[i] = (k < KIN) ? (__bf16)w0[k * 64 + n] : (__bf16)0.0f;
  }
  for (int i = tid; i < 64 * 64; i += 256) sW1[i] = (__bf16)w1[i];
  for (int i = tid; i < 64 * 16; i += 256) {
    const int k = i >> 4, n = i & 15;
    sW2[i] = (n < OUT_W) ? (__bf16)w2[k * OUT_W + n] : (__bf16)0.0f;
  }
  if (tid < 64) { sB0[tid] = b0[tid]; sB1[tid] = b1[tid]; }
  if (tid < 16) sB2[tid] = (tid < OUT_W) ? b2[tid] : 0.0f;
  __syncthreads();

  const int wave = tid >> 5;
  const int lane = tid & 31;
  __bf16* St = &sStage[wave][0];

  // Resident B fragments (full reuse across all tiles handled by this wave).
  v16bf B0[4], B1[2][4], B2f[2];
#pragma unroll
  for (int t = 0; t < 4; ++t) B0[t] = load_b_frag(sW0, 64, 0, t * 16, lane);
#pragma unroll
  for (int ks = 0; ks < 2; ++ks)
#pragma unroll
    for (int t = 0; t < 4; ++t)
      B1[ks][t] = load_b_frag(sW1, 64, ks * 32, t * 16, lane);
#pragma unroll
  for (int ks = 0; ks < 2; ++ks) B2f[ks] = load_b_frag(sW2, 16, ks * 32, 0, lane);

  const int nTiles = (rows + 15) >> 4;
  for (int tile = blockIdx.x * WAVES + wave; tile < nTiles;
       tile += gridDim.x * WAVES) {
    // ---- gather input rows into per-wave LDS staging (bf16, K padded to 32)
    if (lane < 16) {
      const int r = tile * 16 + lane;
#pragma unroll
      for (int c = 0; c < 32; ++c) St[lane * 64 + c] = (__bf16)0.0f;
      if (r < rows) {
        const int ia = idxA ? (int)idxA[r] : r;
#pragma unroll
        for (int c = 0; c < WA; ++c)
          St[lane * 64 + c] = (__bf16)srcA[ia * WA + c];
        if constexpr (WB > 0) {
          const int ib = idxB ? (int)idxB[r] : r;
#pragma unroll
          for (int c = 0; c < WB; ++c)
            St[lane * 64 + WA + c] = (__bf16)srcB[ib * WB + c];
        }
        if constexpr (WC > 0) {
          const int ic = idxC ? (int)idxC[r] : r;
#pragma unroll
          for (int c = 0; c < WC; ++c)
            St[lane * 64 + WA + WB + c] = (__bf16)srcC[ic * WC + c];
        }
      }
    }
    wave_sync_lds();

    // ---- layer 0: [16 x 32] x [32 x 64]  (4 N-tiles, 1 K-slice)
    v16bf A0 = load_a_frag(St, 64, 0, lane);
    v8f h[4];
#pragma unroll
    for (int t = 0; t < 4; ++t) {
      v8f c;
      const float bv = sB0[t * 16 + (lane & 15)];
#pragma unroll
      for (int e = 0; e < 8; ++e) c[e] = bv;
      c = __builtin_amdgcn_wmma_f32_16x16x32_bf16(false, A0, false, B0[t],
                                                  (short)0, c, false, false);
#pragma unroll
      for (int e = 0; e < 8; ++e) c[e] = fmaxf(c[e], 0.0f);
      h[t] = c;
    }

    // ---- repack h (C layout) -> staging [16][64] bf16 for next-layer A
    wave_sync_lds();
#pragma unroll
    for (int t = 0; t < 4; ++t)
#pragma unroll
      for (int e = 0; e < 8; ++e)
        St[((lane < 16) ? e : e + 8) * 64 + t * 16 + (lane & 15)] =
            (__bf16)h[t][e];
    wave_sync_lds();

    // ---- layer 1: [16 x 64] x [64 x 64]  (4 N-tiles, 2 K-slices)
    v16bf A1a = load_a_frag(St, 64, 0, lane);
    v16bf A1b = load_a_frag(St, 64, 32, lane);
#pragma unroll
    for (int t = 0; t < 4; ++t) {
      v8f c;
      const float bv = sB1[t * 16 + (lane & 15)];
#pragma unroll
      for (int e = 0; e < 8; ++e) c[e] = bv;
      c = __builtin_amdgcn_wmma_f32_16x16x32_bf16(false, A1a, false, B1[0][t],
                                                  (short)0, c, false, false);
      c = __builtin_amdgcn_wmma_f32_16x16x32_bf16(false, A1b, false, B1[1][t],
                                                  (short)0, c, false, false);
#pragma unroll
      for (int e = 0; e < 8; ++e) c[e] = fmaxf(c[e], 0.0f);
      h[t] = c;
    }

    wave_sync_lds();
#pragma unroll
    for (int t = 0; t < 4; ++t)
#pragma unroll
      for (int e = 0; e < 8; ++e)
        St[((lane < 16) ? e : e + 8) * 64 + t * 16 + (lane & 15)] =
            (__bf16)h[t][e];
    wave_sync_lds();

    // ---- layer 2: [16 x 64] x [64 x 16]  (1 N-tile, 2 K-slices)
    v16bf A2a = load_a_frag(St, 64, 0, lane);
    v16bf A2b = load_a_frag(St, 64, 32, lane);
    v8f o;
    {
      const float bv = sB2[lane & 15];
#pragma unroll
      for (int e = 0; e < 8; ++e) o[e] = bv;
    }
    o = __builtin_amdgcn_wmma_f32_16x16x32_bf16(false, A2a, false, B2f[0],
                                                (short)0, o, false, false);
    o = __builtin_amdgcn_wmma_f32_16x16x32_bf16(false, A2b, false, B2f[1],
                                                (short)0, o, false, false);

    // ---- epilogue: C element e -> (M = e or e+8, N = lane%16)
    const int n = lane & 15;
    if (n < OUT_W) {
#pragma unroll
      for (int e = 0; e < 8; ++e) {
        const int m = (lane < 16) ? e : e + 8;
        const int r = tile * 16 + m;
        if (r < rows) {
          float v = o[e];
          if constexpr (SIGMOID) v = 1.0f / (1.0f + __expf(-v));
          if constexpr (STORE) out[r * OUT_W + n] = v;
          if constexpr (SCATTER)
            atomicAdd(&agg[(int)idxS[r] * OUT_W + n], v);
        }
      }
    }
  }
}

// ---------------------------------------------------------------------------
// Host launcher
// ---------------------------------------------------------------------------
extern "C" void kernel_launch(void* const* d_in, const int* in_sizes, int n_in,
                              void* d_out, int out_size, void* d_ws,
                              size_t ws_size, hipStream_t stream) {
  const float*     x    = (const float*)d_in[0];
  const long long* eidx = (const long long*)d_in[1];  // int64 [2, E]
  const float*     ea   = (const float*)d_in[2];

  const int N = in_sizes[0] / 3;
  const int E = in_sizes[1] / 2;
  const long long* srcI = eidx;       // edge_index[0]
  const long long* dstI = eidx + E;   // edge_index[1]

  // ---- map the 30 flattened param arrays (5 MLPs x 3 linears x {w,b}).
  // First-layer weight sizes are unique per MLP: 8*64, 9*64, 12*64, 15*64,
  // 21*64 -> assignment is order-independent. Detect b/w ordering from sizes.
  struct Lin { const float *w, *b; };
  struct Mlp { Lin l[3]; };
  Mlp in1_r{}, in1_o{}, in2_r{}, in2_o{}, r2{};
  const int base = 3;
  const bool b_first = (in_sizes[base] <= 64);
  for (int g = 0; g < 5; ++g) {
    const int i0 = base + g * 6;
    Mlp m;
    for (int l = 0; l < 3; ++l) {
      if (b_first) {
        m.l[l].b = (const float*)d_in[i0 + 2 * l + 0];
        m.l[l].w = (const float*)d_in[i0 + 2 * l + 1];
      } else {
        m.l[l].w = (const float*)d_in[i0 + 2 * l + 0];
        m.l[l].b = (const float*)d_in[i0 + 2 * l + 1];
      }
    }
    const int w0sz = in_sizes[b_first ? i0 + 1 : i0 + 0];
    switch (w0sz) {
      case  8 * 64: in1_o = m; break;
      case  9 * 64: in1_r = m; break;
      case 12 * 64: in2_o = m; break;
      case 15 * 64: in2_r = m; break;
      case 21 * 64: r2    = m; break;
      default: break;
    }
  }

  // ---- workspace carve-out (~82 MB)
  char* p = (char*)d_ws;
  auto alloc = [&](size_t bytes) -> void* {
    void* q = (void*)p;
    p += (bytes + 255) & ~(size_t)255;
    return q;
  };
  float* agg1 = (float*)alloc((size_t)N * 5 * sizeof(float));
  float* agg2 = (float*)alloc((size_t)N * 7 * sizeof(float));
  float* x1   = (float*)alloc((size_t)N * 5 * sizeof(float));
  float* x2   = (float*)alloc((size_t)N * 7 * sizeof(float));
  float* e1   = (float*)alloc((size_t)E * 5 * sizeof(float));
  float* e2   = (float*)alloc((size_t)E * 7 * sizeof(float));
  (void)ws_size; (void)n_in; (void)out_size;

  // zero the scatter accumulators every call (covers agg1..agg2 incl. pad)
  {
    const int nz = (int)(((char*)x1 - (char*)agg1) / sizeof(float));
    zero_f32<<<(nz + 255) / 256, 256, 0, stream>>>(agg1, nz);
  }

  auto blocks = [](int rows) {
    const int tiles = (rows + 15) >> 4;
    return (tiles + WAVES - 1) / WAVES;
  };

  // IN1 relational: [x[dst] | x[src] | edge_attr] (9) -> e1 [E,5]; agg1 += e1 @ dst
  mlp_tile_kernel<3, 3, 3, 5, false, true, true>
      <<<blocks(E), 256, 0, stream>>>(
          x, dstI, x, srcI, ea, nullptr,
          in1_r.l[0].w, in1_r.l[0].b, in1_r.l[1].w, in1_r.l[1].b,
          in1_r.l[2].w, in1_r.l[2].b,
          e1, agg1, dstI, E);

  // IN1 object: [x | agg1] (8) -> x1 [N,5]
  mlp_tile_kernel<3, 5, 0, 5, false, true, false>
      <<<blocks(N), 256, 0, stream>>>(
          x, nullptr, agg1, nullptr, nullptr, nullptr,
          in1_o.l[0].w, in1_o.l[0].b, in1_o.l[1].w, in1_o.l[1].b,
          in1_o.l[2].w, in1_o.l[2].b,
          x1, nullptr, nullptr, N);

  // IN2 relational: [x1[dst] | x1[src] | e1] (15) -> e2 [E,7]; agg2 += e2 @ dst
  mlp_tile_kernel<5, 5, 5, 7, false, true, true>
      <<<blocks(E), 256, 0, stream>>>(
          x1, dstI, x1, srcI, e1, nullptr,
          in2_r.l[0].w, in2_r.l[0].b, in2_r.l[1].w, in2_r.l[1].b,
          in2_r.l[2].w, in2_r.l[2].b,
          e2, agg2, dstI, E);

  // IN2 object: [x1 | agg2] (12) -> x2 [N,7]
  mlp_tile_kernel<5, 7, 0, 7, false, true, false>
      <<<blocks(N), 256, 0, stream>>>(
          x1, nullptr, agg2, nullptr, nullptr, nullptr,
          in2_o.l[0].w, in2_o.l[0].b, in2_o.l[1].w, in2_o.l[1].b,
          in2_o.l[2].w, in2_o.l[2].b,
          x2, nullptr, nullptr, N);

  // final classifier: [x2[dst] | x2[src] | e2] (21) -> sigmoid -> d_out [E,1]
  mlp_tile_kernel<7, 7, 7, 1, true, true, false>
      <<<blocks(E), 256, 0, stream>>>(
          x2, dstI, x2, srcI, e2, nullptr,
          r2.l[0].w, r2.l[0].b, r2.l[1].w, r2.l[1].b,
          r2.l[2].w, r2.l[2].b,
          (float*)d_out, nullptr, nullptr, E);
}
